// MultiHeadAttention_75350906241780
// MI455X (gfx1250) — compile-verified, matrix operand order checked by reading, not run on hardware
//
#include <hip/hip_runtime.h>
#include <hip/hip_bf16.h>

typedef __attribute__((ext_vector_type(16))) __bf16 v16bf;
typedef __attribute__((ext_vector_type(8)))  float  v8f;

union ABu { v16bf v; unsigned int u[8]; };
union Cu  { v8f   v; float        f[8]; };

#define SEQ    2048
#define DMODEL 1024
#define HEADS  16
#define HD     64
#define MTOT   (2 * SEQ)   // 4096 rows (b*n)

__device__ __forceinline__ unsigned short f2bf(float f) {
    unsigned int u = __float_as_uint(f);
    unsigned int r = (u + 0x7FFFu + ((u >> 16) & 1u)) >> 16;
    return (unsigned short)r;
}

__device__ __forceinline__ int kpack(int p, int half) {
    // 16-bit A/B operand K packing per CDNA5 ISA 7.12.2 (16x32 tile)
    return (p < 4) ? (2 * p + 8 * half) : (16 + 2 * (p - 4) + 8 * half);
}

// load one 16x32 bf16 WMMA operand for this lane; p0 = row base + k-offset
__device__ __forceinline__ void load16x32(ABu& d, const unsigned short* p0, int half) {
#pragma unroll
    for (int p = 0; p < 8; p++)
        d.u[p] = *(const unsigned int*)(p0 + kpack(p, half));
}

// ---------------- elementwise converts ----------------
__global__ void cvt_f32_bf16(const float* __restrict__ in,
                             unsigned short* __restrict__ out, int n) {
    int i = blockIdx.x * blockDim.x + threadIdx.x;
    if (i < n) out[i] = f2bf(in[i]);
}

__global__ void transpose_cvt(const float* __restrict__ W,
                              unsigned short* __restrict__ WT, int K, int N) {
    int i = blockIdx.x * blockDim.x + threadIdx.x;
    if (i >= K * N) return;
    int k = i / N, n = i % N;
    WT[(size_t)n * K + k] = f2bf(W[(size_t)k * N + n]);
}

// ------- bf16 WMMA GEMM: wave computes 32(M) x 64(N), double-buffered operands -------
// MODE 0: out bf16 [b,h,n,d]   (Q, K)
// MODE 1: out bf16 [b,h,d,n]   (V transposed)
// MODE 2: out f32  [m, N] + bias
template<int MODE>
__global__ void __launch_bounds__(256) gemm32x64(
    const unsigned short* __restrict__ A,   // MTOT x K row-major bf16
    const unsigned short* __restrict__ BT,  // N x K row-major bf16 (W^T)
    void* __restrict__ outp, const float* __restrict__ bias)
{
    constexpr int K  = DMODEL;
    constexpr int NT = DMODEL / 64;          // 16 n-tiles
    const int lane = threadIdx.x & 31;
    const int wave = threadIdx.x >> 5;
    const int tile = blockIdx.x * 8 + wave;  // (MTOT/32)*(NT) = 2048 tiles
    const int mbase = (tile / NT) * 32;
    const int nbase = (tile % NT) * 64;
    const int half = lane >> 4;
    const int l16  = lane & 15;

    const unsigned short* arow[2];
    arow[0] = A + (size_t)(mbase + l16) * K;
    arow[1] = A + (size_t)(mbase + 16 + l16) * K;
    const unsigned short* brow[4];
#pragma unroll
    for (int t = 0; t < 4; t++)
        brow[t] = BT + (size_t)(nbase + t * 16 + l16) * K;

    Cu acc[2][4];
#pragma unroll
    for (int s = 0; s < 2; s++)
#pragma unroll
        for (int t = 0; t < 4; t++)
#pragma unroll
            for (int r = 0; r < 8; r++) acc[s][t].f[r] = 0.f;

    // modulo-2 register double buffers: loads for step i+1 issue before WMMAs of step i
    ABu ab[2][2], bb[2][4];
#pragma unroll
    for (int s = 0; s < 2; s++) load16x32(ab[0][s], arow[s], half);
#pragma unroll
    for (int t = 0; t < 4; t++) load16x32(bb[0][t], brow[t], half);

#pragma unroll
    for (int it = 0; it < K / 32; ++it) {
        const int cur = it & 1, nxt = cur ^ 1;
        if (it + 1 < K / 32) {
            const int ks = (it + 1) * 32;
#pragma unroll
            for (int s = 0; s < 2; s++) load16x32(ab[nxt][s], arow[s] + ks, half);
#pragma unroll
            for (int t = 0; t < 4; t++) load16x32(bb[nxt][t], brow[t] + ks, half);
        }
#pragma unroll
        for (int s = 0; s < 2; s++)
#pragma unroll
            for (int t = 0; t < 4; t++)
                acc[s][t].v = __builtin_amdgcn_wmma_f32_16x16x32_bf16(
                    false, ab[cur][s].v, false, bb[cur][t].v,
                    (short)0, acc[s][t].v, false, false);
    }

    if (MODE == 2) {
        float* O = (float*)outp;
#pragma unroll
        for (int t = 0; t < 4; t++) {
            const int col = nbase + t * 16 + l16;
            const float bv = bias[col];
#pragma unroll
            for (int s = 0; s < 2; s++)
#pragma unroll
                for (int r = 0; r < 8; r++) {
                    const int row = mbase + s * 16 + r + 8 * half;
                    O[(size_t)row * DMODEL + col] = acc[s][t].f[r] + bv;
                }
        }
    } else {
        unsigned short* O = (unsigned short*)outp;
#pragma unroll
        for (int t = 0; t < 4; t++) {
            const int col = nbase + t * 16 + l16;
            const int h = col >> 6, d = col & 63;
#pragma unroll
            for (int s = 0; s < 2; s++)
#pragma unroll
                for (int r = 0; r < 8; r++) {
                    const int row = mbase + s * 16 + r + 8 * half;
                    const int bb2 = row >> 11, sq = row & (SEQ - 1);
                    const size_t off = (MODE == 0)
                        ? ((size_t)((bb2 * HEADS + h) * SEQ + sq) * HD + d)
                        : ((size_t)((bb2 * HEADS + h) * HD + d) * SEQ + sq);
                    O[off] = f2bf(acc[s][t].f[r]);
                }
        }
    }
}

// ---------------- causal flash attention: one wave per 16-query tile ----------------
__global__ void __launch_bounds__(128) attn_kernel(
    const unsigned short* __restrict__ Q,   // [b,h,n,d] bf16
    const unsigned short* __restrict__ Km,  // [b,h,n,d] bf16
    const unsigned short* __restrict__ Vt,  // [b,h,d,n] bf16
    unsigned short* __restrict__ ctx)       // [b*n, 1024] bf16
{
    __shared__ unsigned short Plds[4][16 * 32];
    const int lane = threadIdx.x & 31;
    const int wave = threadIdx.x >> 5;
    const int wg = blockIdx.x * 4 + wave;
    const int qt = wg & 127;
    const int h  = (wg >> 7) & (HEADS - 1);
    const int b  = wg >> 11;
    const int qbase = qt * 16;
    const int half = lane >> 4;
    const int l16  = lane & 15;
    const float scale = 0.125f;  // 1/sqrt(64)

    const unsigned short* Qh = Q  + (size_t)(b * HEADS + h) * SEQ * HD;
    const unsigned short* Kh = Km + (size_t)(b * HEADS + h) * SEQ * HD;
    const unsigned short* Vh = Vt + (size_t)(b * HEADS + h) * HD * SEQ;

    // Q A-operands: two 16x32 chunks over head dim
    ABu qa[2];
#pragma unroll
    for (int c = 0; c < 2; c++)
        load16x32(qa[c], Qh + (size_t)(qbase + l16) * HD + c * 32, half);

    Cu o[4];
#pragma unroll
    for (int t = 0; t < 4; t++)
#pragma unroll
        for (int r = 0; r < 8; r++) o[t].f[r] = 0.f;
    float mrow[8], lrow[8];
#pragma unroll
    for (int r = 0; r < 8; r++) { mrow[r] = -1e30f; lrow[r] = 0.f; }

    const int nchunks = (qbase + 16 + 31) >> 5;   // causal: only up to diagonal
    for (int ch = 0; ch < nchunks; ++ch) {
        const int kb32 = ch * 32;

        // issue ALL global loads for this chunk up front: K tiles + V tiles.
        // Their latency overlaps the softmax VALU work below.
        ABu kop[2][2];   // [t2][head-dim chunk]
#pragma unroll
        for (int t2 = 0; t2 < 2; t2++) {
            const unsigned short* kr = Kh + (size_t)(kb32 + t2 * 16 + l16) * HD;
            load16x32(kop[t2][0], kr, half);
            load16x32(kop[t2][1], kr + 32, half);
        }
        ABu vb[4];
#pragma unroll
        for (int t = 0; t < 4; t++)
            load16x32(vb[t], Vh + (size_t)(t * 16 + l16) * SEQ + kb32, half);
        if (ch + 1 < nchunks) {   // prefetch next chunk's K/V cachelines
            __builtin_prefetch(Kh + (size_t)(kb32 + 32 + l16) * HD, 0, 3);
            __builtin_prefetch(Vh + (size_t)(l16) * SEQ + kb32 + 32, 0, 3);
        }

        // two 16x16 score tiles (S = Q K^T), each = 2 chained WMMAs over head dim
        float sv[2][8], mnew[8];
#pragma unroll
        for (int t2 = 0; t2 < 2; t2++) {
            Cu s;
#pragma unroll
            for (int r = 0; r < 8; r++) s.f[r] = 0.f;
            s.v = __builtin_amdgcn_wmma_f32_16x16x32_bf16(
                false, qa[0].v, false, kop[t2][0].v, (short)0, s.v, false, false);
            s.v = __builtin_amdgcn_wmma_f32_16x16x32_bf16(
                false, qa[1].v, false, kop[t2][1].v, (short)0, s.v, false, false);
            const int col = kb32 + t2 * 16 + l16;
#pragma unroll
            for (int r = 0; r < 8; r++) {
                const int qrow = qbase + r + 8 * half;
                sv[t2][r] = (col <= qrow) ? s.f[r] * scale : -1e30f;
            }
        }

        // online softmax: row max over 32 keys (reduce across 16-lane half)
#pragma unroll
        for (int r = 0; r < 8; r++) {
            float tm = fmaxf(sv[0][r], sv[1][r]);
            tm = fmaxf(tm, __shfl_xor(tm, 1, 16));
            tm = fmaxf(tm, __shfl_xor(tm, 2, 16));
            tm = fmaxf(tm, __shfl_xor(tm, 4, 16));
            tm = fmaxf(tm, __shfl_xor(tm, 8, 16));
            const float mn = fmaxf(mrow[r], tm);
            const float rs = __expf(mrow[r] - mn);
            mrow[r] = mn; mnew[r] = mn;
            lrow[r] *= rs;
#pragma unroll
            for (int t = 0; t < 4; t++) o[t].f[r] *= rs;
        }

        // P = exp(S - m); C-layout -> LDS (row-major 16x32 bf16)
#pragma unroll
        for (int t2 = 0; t2 < 2; t2++)
#pragma unroll
            for (int r = 0; r < 8; r++) {
                const float p = __expf(sv[t2][r] - mnew[r]);
                sv[t2][r] = p;
                Plds[wave][(r + 8 * half) * 32 + t2 * 16 + l16] = f2bf(p);
            }
#pragma unroll
        for (int r = 0; r < 8; r++) {
            float ps = sv[0][r] + sv[1][r];
            ps += __shfl_xor(ps, 1, 16);
            ps += __shfl_xor(ps, 2, 16);
            ps += __shfl_xor(ps, 4, 16);
            ps += __shfl_xor(ps, 8, 16);
            lrow[r] += ps;
        }

        // same-wave LDS ordering (no block barrier: waves have divergent trip counts)
        asm volatile("s_wait_dscnt 0" ::: "memory");

        // reload P in A-layout, then ctx += P(16x32) x V(32x64): 4 WMMAs
        ABu pa;
#pragma unroll
        for (int p = 0; p < 8; p++)
            pa.u[p] = *(const unsigned int*)(&Plds[wave][l16 * 32 + kpack(p, half)]);
#pragma unroll
        for (int t = 0; t < 4; t++)
            o[t].v = __builtin_amdgcn_wmma_f32_16x16x32_bf16(
                false, pa.v, false, vb[t].v, (short)0, o[t].v, false, false);
    }

    // normalize, store ctx bf16 row-major [b*SEQ, 1024]
#pragma unroll
    for (int r = 0; r < 8; r++) {
        const float inv = 1.0f / lrow[r];
        const int qrow = qbase + r + 8 * half;
        const size_t rowoff = ((size_t)b * SEQ + qrow) * DMODEL + h * HD;
#pragma unroll
        for (int t = 0; t < 4; t++)
            ctx[rowoff + t * 16 + l16] = f2bf(o[t].f[r] * inv);
    }
}

extern "C" void kernel_launch(void* const* d_in, const int* in_sizes, int n_in,
                              void* d_out, int out_size, void* d_ws, size_t ws_size,
                              hipStream_t stream) {
    const float* X  = (const float*)d_in[0];
    const float* Wq = (const float*)d_in[1];
    const float* Wk = (const float*)d_in[2];
    const float* Wv = (const float*)d_in[3];
    const float* Wo = (const float*)d_in[4];
    const float* bo = (const float*)d_in[5];
    float* out = (float*)d_out;

    char* ws = (char*)d_ws;
    const size_t MB = 1u << 20;
    unsigned short* Xbf = (unsigned short*)(ws);              //  8 MiB: 4096x1024 bf16
    unsigned short* WqT = (unsigned short*)(ws + 8  * MB);    //  2 MiB each
    unsigned short* WkT = (unsigned short*)(ws + 10 * MB);
    unsigned short* WvT = (unsigned short*)(ws + 12 * MB);
    unsigned short* WoT = (unsigned short*)(ws + 14 * MB);
    unsigned short* Qb  = (unsigned short*)(ws + 16 * MB);    //  8 MiB [b,h,n,d]
    unsigned short* Kb  = (unsigned short*)(ws + 24 * MB);    //  8 MiB [b,h,n,d]
    unsigned short* Vtb = (unsigned short*)(ws + 32 * MB);    //  8 MiB [b,h,d,n]
    unsigned short* Ctx = (unsigned short*)(ws + 40 * MB);    //  8 MiB [b*n,1024]

    const int nX = MTOT * DMODEL;
    cvt_f32_bf16<<<(nX + 255) / 256, 256, 0, stream>>>(X, Xbf, nX);
    const int nW = DMODEL * DMODEL;
    transpose_cvt<<<(nW + 255) / 256, 256, 0, stream>>>(Wq, WqT, DMODEL, DMODEL);
    transpose_cvt<<<(nW + 255) / 256, 256, 0, stream>>>(Wk, WkT, DMODEL, DMODEL);
    transpose_cvt<<<(nW + 255) / 256, 256, 0, stream>>>(Wv, WvT, DMODEL, DMODEL);
    transpose_cvt<<<(nW + 255) / 256, 256, 0, stream>>>(Wo, WoT, DMODEL, DMODEL);

    // (MTOT/32)*(DMODEL/64) = 2048 wave-tiles / 8 waves per block = 256 blocks
    gemm32x64<0><<<256, 256, 0, stream>>>(Xbf, WqT, Qb,  nullptr);
    gemm32x64<0><<<256, 256, 0, stream>>>(Xbf, WkT, Kb,  nullptr);
    gemm32x64<1><<<256, 256, 0, stream>>>(Xbf, WvT, Vtb, nullptr);

    // 2 batches * 16 heads * 128 q-tiles = 4096 waves / 4 per block
    attn_kernel<<<1024, 128, 0, stream>>>(Qb, Kb, Vtb, Ctx);

    gemm32x64<2><<<256, 256, 0, stream>>>(Ctx, WoT, out, bo);
}
